// FARGAN_8435315769400
// MI455X (gfx1250) — matrix operand order repaired and checked by reading, not compiled, not run on hardware
//
#include <hip/hip_runtime.h>
#include <hip/hip_bf16.h>
#include <stdint.h>

// ---------------------------------------------------------------------------
// Types for CDNA5 WMMA (wave32): D(16x16 f32) = A(16x32 f16) x B(32x16 f16) + C
// ---------------------------------------------------------------------------
typedef __attribute__((ext_vector_type(16))) _Float16 v16h;
typedef __attribute__((ext_vector_type(8)))  _Float16 v8h;
typedef __attribute__((ext_vector_type(8)))  float    v8f;

#define SDIM      64
#define TFRAMES   100
#define HOPLEN    256
#define ROWS      16      // batch rows per workgroup (one WMMA M-tile)
#define NTHREADS  128     // 4 wave32s

// ---- packed f16 weight offsets in workspace (units: halves) ----------------
// Packed layout per matrix Wt (K x N): for tile nt (N/16), kb (K/32), lane(0..31),
// e(0..15): value = W[o*K + i] with o = nt*16 + (lane&15), i = kb*32 + (lane>>4)*16 + e.
// Each lane reads 16 contiguous halves (32B) per K-step -> two b128 loads.
static constexpr size_t OFF_W1    = 0;
static constexpr size_t OFF_W2    = OFF_W1    + 512ull * 512;   // in_W1  K=512 N=512
static constexpr size_t OFF_FW    = OFF_W2    + 512ull * 256;   // in_W2  K=512 N=256
static constexpr size_t OFF_FWGLU = OFF_FW    + 128ull * 64;    // fw_W   K=128 N=64
static constexpr size_t OFF_WIH1  = OFF_FWGLU + 64ull  * 64;
static constexpr size_t OFF_WHH1  = OFF_WIH1  + 128ull * 192;
static constexpr size_t OFF_GLU1  = OFF_WHH1  + 64ull  * 192;
static constexpr size_t OFF_WIH2  = OFF_GLU1  + 64ull  * 64;
static constexpr size_t OFF_WHH2  = OFF_WIH2  + 128ull * 192;
static constexpr size_t OFF_GLU2  = OFF_WHH2  + 64ull  * 192;
static constexpr size_t OFF_WIH3  = OFF_GLU2  + 64ull  * 64;
static constexpr size_t OFF_WHH3  = OFF_WIH3  + 128ull * 192;
static constexpr size_t OFF_GLU3  = OFF_WHH3  + 64ull  * 192;
static constexpr size_t OFF_SKIPD = OFF_GLU3  + 64ull  * 64;    // K=320 N=128
static constexpr size_t OFF_SKIPG = OFF_SKIPD + 320ull * 128;   // K=128 N=128
static constexpr size_t OFF_OUTW  = OFF_SKIPG + 128ull * 128;   // K=128 N=64
static constexpr size_t OFF_END   = OFF_OUTW  + 128ull * 64;

// ---------------------------------------------------------------------------
__device__ __forceinline__ float sigmoidf_(float x) {
    return 1.0f / (1.0f + __expf(-x));
}

__device__ __forceinline__ uint32_t mix32(uint32_t x) {
    x ^= x >> 16; x *= 0x7feb352dU;
    x ^= x >> 15; x *= 0x846ca68bU;
    x ^= x >> 16; return x;
}

// Deterministic per-element uniform noise: clip(x + (u-0.5)/127, -1, 1)
__device__ __forceinline__ float noisef(float x, uint32_t step, uint32_t salt, uint32_t idx) {
    uint32_t h = mix32((step * 0x9E3779B9u) ^ ((salt + 0x51eDu) * 0x85EBCA6Bu) ^ (idx * 0xC2B2AE35u));
    float u = (float)(h >> 8) * (1.0f / 16777216.0f) - 0.5f;
    float y = fmaf(u, 1.0f / 127.0f, x);
    return fminf(1.0f, fmaxf(-1.0f, y));
}

__device__ __forceinline__ v8f vzero8() {
    v8f z;
#pragma unroll
    for (int i = 0; i < 8; ++i) z[i] = 0.0f;
    return z;
}

// ---------------------------------------------------------------------------
// One 16x16 output tile: C += A(16 x 32*ktiles, LDS f16 row-major, stride lda)
//                           x Bpacked(tile nt, ktiles K-blocks, global f16)
// A and B fragments both use contiguous k = 16*(lane/16) + e within each 32-wide K-block.
// ---------------------------------------------------------------------------
__device__ __forceinline__ v8f wmma_tile(const _Float16* A, int lda,
                                         const _Float16* __restrict__ Bp,
                                         int ktiles, int nt, int lane, v8f acc) {
    const int hi = lane >> 4;
    const int lo = lane & 15;
    const _Float16* arow = A + (size_t)lo * lda + hi * 16;
    const _Float16* bb   = Bp + (size_t)nt * ktiles * 512 + lane * 16;
    for (int kb = 0; kb < ktiles; ++kb) {
        v8h a0 = *(const v8h*)(arow + kb * 32);        // K = hi*16 + e
        v8h a1 = *(const v8h*)(arow + kb * 32 + 8);    // K = hi*16 + 8 + e
        v8h b0 = *(const v8h*)(bb + (size_t)kb * 512);
        v8h b1 = *(const v8h*)(bb + (size_t)kb * 512 + 8);
        v16h a = __builtin_shufflevector(a0, a1, 0,1,2,3,4,5,6,7,8,9,10,11,12,13,14,15);
        v16h b = __builtin_shufflevector(b0, b1, 0,1,2,3,4,5,6,7,8,9,10,11,12,13,14,15);
        acc = __builtin_amdgcn_wmma_f32_16x16x32_f16(false, a, false, b, (short)0, acc,
                                                     false, false);
    }
    return acc;
}

// ---------------------------------------------------------------------------
// Weight packing: f32 (N x K row-major, as W[o][i]) -> packed f16 fragments.
// ---------------------------------------------------------------------------
__global__ void fargan_pack_w(const float* __restrict__ W, _Float16* __restrict__ out,
                              int K, int N) {
    int total = K * N;
    int kt = K >> 5;
    for (int idx = blockIdx.x * blockDim.x + threadIdx.x; idx < total;
         idx += gridDim.x * blockDim.x) {
        int e    = idx & 15;
        int lane = (idx >> 4) & 31;
        int rest = idx >> 9;
        int kb   = rest % kt;
        int nt   = rest / kt;
        int o = nt * 16 + (lane & 15);
        int i = kb * 32 + (lane >> 4) * 16 + e;
        out[idx] = (_Float16)W[(size_t)o * K + i];
    }
}

// ---------------------------------------------------------------------------
// GRU cell + GLU layer.
//  pA: cat input [x, prevN] (K=128, f16). pB: scratch f16 A buffer (stride 512).
//  On exit pG holds g (f16); if buildNext, pA holds [g, prevN] for the next GRU.
// ---------------------------------------------------------------------------
__device__ void gru_glu_layer(int tid, int wave, int lane, int m0, uint32_t step,
                              uint32_t saltA, uint32_t saltB,
                              const _Float16* __restrict__ wih,
                              const _Float16* __restrict__ whh,
                              const _Float16* __restrict__ wglu,
                              _Float16 (*pA)[512], _Float16 (*pB)[512],
                              float (*pH)[SDIM], float (*pT0)[192], float (*pT1)[192],
                              _Float16 (*pPrevN)[SDIM], _Float16 (*pG)[SDIM],
                              bool buildNext) {
    const int hi = lane >> 4;
    const int lo = lane & 15;

    // stage h as f16 A fragment source
    for (int i = tid; i < ROWS * SDIM; i += NTHREADS) {
        int r = i >> 6, c = i & 63;
        pB[r][c] = (_Float16)pH[r][c];
    }
    __syncthreads();

    // gi = cat @ Wih^T (N=192, K=128); gh = h @ Whh^T (N=192, K=64)
    for (int q = 0; q < 3; ++q) {
        int nt = wave + 4 * q;
        v8f ai = wmma_tile(&pA[0][0], 512, wih, 4, nt, lane, vzero8());
        v8f ah = wmma_tile(&pB[0][0], 512, whh, 2, nt, lane, vzero8());
        int n = nt * 16 + lo;
#pragma unroll
        for (int r2 = 0; r2 < 8; ++r2) {
            int m = hi * 8 + r2;
            pT0[m][n] = ai[r2];
            pT1[m][n] = ah[r2];
        }
    }
    __syncthreads();

    // GRU combine + pre-GLU noise
    for (int i = tid; i < ROWS * SDIM; i += NTHREADS) {
        int r = i >> 6, c = i & 63;
        float rg = sigmoidf_(pT0[r][c]       + pT1[r][c]);
        float z  = sigmoidf_(pT0[r][64 + c]  + pT1[r][64 + c]);
        float nn = tanhf(pT0[r][128 + c] + rg * pT1[r][128 + c]);
        float h  = pH[r][c];
        float hnew = (1.0f - z) * nn + z * h;
        pH[r][c] = hnew;
        uint32_t gid = (uint32_t)((m0 + r) * SDIM + c);
        float xg = noisef(hnew, step, saltA, gid);
        pT0[r][c] = xg;                 // keep f32 for GLU product
        pB[r][c]  = (_Float16)xg;       // A fragment source for GLU matmul
    }
    __syncthreads();

    // GLU: g = noise(x * sigmoid(x @ Wglu^T))  (N=64, K=64 -> 1 tile per wave)
    {
        v8f acc = wmma_tile(&pB[0][0], 512, wglu, 2, wave, lane, vzero8());
        int n = wave * 16 + lo;
#pragma unroll
        for (int r2 = 0; r2 < 8; ++r2) {
            int m = hi * 8 + r2;
            float g = noisef(pT0[m][n] * sigmoidf_(acc[r2]), step, saltB,
                             (uint32_t)((m0 + m) * SDIM + n));
            pG[m][n] = (_Float16)g;
            if (buildNext) {
                pA[m][n]      = (_Float16)g;
                pA[m][64 + n] = pPrevN[m][n];
            }
        }
    }
    __syncthreads();
}

// ---------------------------------------------------------------------------
// Main persistent kernel: 4 workgroups x 16 batch rows, full T x NSUB recurrence.
// ---------------------------------------------------------------------------
__global__ __launch_bounds__(NTHREADS, 1)
void fargan_main(const float* __restrict__ features,   // (64, 256, 100)
                 const float* __restrict__ gfeat,      // (64, 256, 1)
                 const float* __restrict__ b1,         // (512,)
                 const float* __restrict__ b2,         // (256,)
                 const _Float16* __restrict__ wp,      // packed weights
                 float* __restrict__ out) {            // (64, 1, 25600)
    __shared__ __align__(16) _Float16 sA[ROWS][512];   // A staging (cat inputs)
    __shared__ __align__(16) _Float16 sB[ROWS][512];   // X1 / generic f16 A scratch
    __shared__ float    sX[ROWS][256];                 // frame subframe source x
    __shared__ float    sT0[ROWS][192];                // f32 staging (gi / glu-x)
    __shared__ float    sT1[ROWS][192];                // f32 staging (gh)
    __shared__ float    sH[3][ROWS][SDIM];             // GRU states
    __shared__ float    sPrev[ROWS][SDIM];             // previous subframe output
    __shared__ __align__(16) _Float16 sPrevN[ROWS][SDIM]; // noised prev (f16)
    __shared__ __align__(16) _Float16 sS3[ROWS][SDIM];    // s3 state (noised sub)
    __shared__ __align__(16) _Float16 sSub[ROWS][SDIM];   // this subframe's noised sub
    __shared__ __align__(16) _Float16 sFw[ROWS][SDIM];    // fw (f16, for skip cat)
    __shared__ __align__(16) _Float16 sG[3][ROWS][SDIM];  // g1,g2,g3 (f16)

    const int tid  = threadIdx.x;
    const int wave = tid >> 5;
    const int lane = tid & 31;
    const int hi   = lane >> 4;
    const int lo   = lane & 15;
    const int m0   = blockIdx.x * ROWS;

    // zero-init recurrent state
    for (int i = tid; i < ROWS * SDIM; i += NTHREADS) {
        int r = i >> 6, c = i & 63;
        sH[0][r][c] = 0.0f; sH[1][r][c] = 0.0f; sH[2][r][c] = 0.0f;
        sPrev[r][c] = 0.0f; sS3[r][c] = (_Float16)0.0f;
    }
    __syncthreads();

    for (int t = 0; t < TFRAMES; ++t) {
        // F1: build cat(features[:,:,t], g) as f16 A
        for (int i = tid; i < ROWS * 512; i += NTHREADS) {
            int r = i >> 9, k = i & 511;
            float v = (k < 256)
                        ? features[((size_t)(m0 + r) * 256 + k) * TFRAMES + t]
                        : gfeat[(size_t)(m0 + r) * 256 + (k - 256)];
            sA[r][k] = (_Float16)v;
        }
        __syncthreads();

        // F2: X1 = tanh(cat @ in_W1^T + b1)   (N=512, K=512)
        for (int q = 0; q < 8; ++q) {
            int nt = wave + 4 * q;
            v8f acc = wmma_tile(&sA[0][0], 512, wp + OFF_W1, 16, nt, lane, vzero8());
            int n = nt * 16 + lo;
            float bias = b1[n];
#pragma unroll
            for (int r2 = 0; r2 < 8; ++r2) {
                int m = hi * 8 + r2;
                sB[m][n] = (_Float16)tanhf(acc[r2] + bias);
            }
        }
        __syncthreads();

        // F3: x = X1 @ in_W2^T + b2   (N=256, K=512)
        for (int q = 0; q < 4; ++q) {
            int nt = wave + 4 * q;
            v8f acc = wmma_tile(&sB[0][0], 512, wp + OFF_W2, 16, nt, lane, vzero8());
            int n = nt * 16 + lo;
            float bias = b2[n];
#pragma unroll
            for (int r2 = 0; r2 < 8; ++r2) {
                int m = hi * 8 + r2;
                sX[m][n] = acc[r2] + bias;
            }
        }
        __syncthreads();

        for (int s = 0; s < 4; ++s) {
            const uint32_t step = (uint32_t)(t * 4 + s);

            // noise sub & prev; build cat(sub_n, s3) as fw input
            for (int i = tid; i < ROWS * SDIM; i += NTHREADS) {
                int r = i >> 6, c = i & 63;
                uint32_t gid = (uint32_t)((m0 + r) * SDIM + c);
                float sub = noisef(sX[r][s * 64 + c], step, 0, gid);
                float pn  = noisef(sPrev[r][c],       step, 1, gid);
                sSub[r][c]   = (_Float16)sub;
                sA[r][c]     = (_Float16)sub;
                sA[r][64 + c] = sS3[r][c];
                sPrevN[r][c] = (_Float16)pn;
            }
            __syncthreads();

            // fw dense: tanh(cat @ fw_W^T)   (N=64, K=128)
            {
                v8f acc = wmma_tile(&sA[0][0], 512, wp + OFF_FW, 4, wave, lane, vzero8());
                int n = wave * 16 + lo;
#pragma unroll
                for (int r2 = 0; r2 < 8; ++r2) {
                    int m = hi * 8 + r2;
                    float st = tanhf(acc[r2]);
                    sT0[m][n] = st;
                    sB[m][n]  = (_Float16)st;
                }
            }
            __syncthreads();

            // fw GLU + noise; build cat(fw, prevN) for GRU1
            {
                v8f acc = wmma_tile(&sB[0][0], 512, wp + OFF_FWGLU, 2, wave, lane, vzero8());
                int n = wave * 16 + lo;
#pragma unroll
                for (int r2 = 0; r2 < 8; ++r2) {
                    int m = hi * 8 + r2;
                    float fw = noisef(sT0[m][n] * sigmoidf_(acc[r2]), step, 2,
                                      (uint32_t)((m0 + m) * SDIM + n));
                    sFw[m][n]    = (_Float16)fw;
                    sA[m][n]     = (_Float16)fw;
                    sA[m][64 + n] = sPrevN[m][n];
                }
            }
            __syncthreads();

            // three GRU+GLU layers
            gru_glu_layer(tid, wave, lane, m0, step, 3, 4,
                          wp + OFF_WIH1, wp + OFF_WHH1, wp + OFF_GLU1,
                          sA, sB, sH[0], sT0, sT1, sPrevN, sG[0], true);
            gru_glu_layer(tid, wave, lane, m0, step, 5, 6,
                          wp + OFF_WIH2, wp + OFF_WHH2, wp + OFF_GLU2,
                          sA, sB, sH[1], sT0, sT1, sPrevN, sG[1], true);
            gru_glu_layer(tid, wave, lane, m0, step, 7, 8,
                          wp + OFF_WIH3, wp + OFF_WHH3, wp + OFF_GLU3,
                          sA, sB, sH[2], sT0, sT1, sPrevN, sG[2], false);

            // skip concat [g1,g2,g3,fw,prevN]  (K=320)
            for (int i = tid; i < ROWS * 320; i += NTHREADS) {
                int r = i / 320, k = i % 320;
                _Float16 v;
                if      (k < 64)  v = sG[0][r][k];
                else if (k < 128) v = sG[1][r][k - 64];
                else if (k < 192) v = sG[2][r][k - 128];
                else if (k < 256) v = sFw[r][k - 192];
                else              v = sPrevN[r][k - 256];
                sA[r][k] = v;
            }
            __syncthreads();

            // skip dense: noise(tanh(feats @ skip_dense^T))   (N=128, K=320)
            for (int q = 0; q < 2; ++q) {
                int nt = wave + 4 * q;
                v8f acc = wmma_tile(&sA[0][0], 512, wp + OFF_SKIPD, 10, nt, lane, vzero8());
                int n = nt * 16 + lo;
#pragma unroll
                for (int r2 = 0; r2 < 8; ++r2) {
                    int m = hi * 8 + r2;
                    float x = noisef(tanhf(acc[r2]), step, 9,
                                     (uint32_t)((m0 + m) * 128 + n));
                    sT0[m][n] = x;
                    sB[m][n]  = (_Float16)x;
                }
            }
            __syncthreads();

            // skip GLU   (N=128, K=128)
            for (int q = 0; q < 2; ++q) {
                int nt = wave + 4 * q;
                v8f acc = wmma_tile(&sB[0][0], 512, wp + OFF_SKIPG, 4, nt, lane, vzero8());
                int n = nt * 16 + lo;
#pragma unroll
                for (int r2 = 0; r2 < 8; ++r2) {
                    int m = hi * 8 + r2;
                    float sk = sT0[m][n] * sigmoidf_(acc[r2]);
                    sA[m][n] = (_Float16)sk;
                }
            }
            __syncthreads();

            // out = tanh(skip @ out_W^T)   (N=64, K=128); store & update prev
            {
                v8f acc = wmma_tile(&sA[0][0], 512, wp + OFF_OUTW, 4, wave, lane, vzero8());
                int n = wave * 16 + lo;
#pragma unroll
                for (int r2 = 0; r2 < 8; ++r2) {
                    int m = hi * 8 + r2;
                    float o = tanhf(acc[r2]);
                    sPrev[m][n] = o;
                    out[(size_t)(m0 + m) * (TFRAMES * HOPLEN) + t * HOPLEN + s * 64 + n] = o;
                }
            }
            __syncthreads();

            // s3 <- this subframe's noised sub
            for (int i = tid; i < ROWS * SDIM; i += NTHREADS) {
                int r = i >> 6, c = i & 63;
                sS3[r][c] = sSub[r][c];
            }
            __syncthreads();
        }
    }
}

// ---------------------------------------------------------------------------
extern "C" void kernel_launch(void* const* d_in, const int* in_sizes, int n_in,
                              void* d_out, int out_size, void* d_ws, size_t ws_size,
                              hipStream_t stream) {
    (void)in_sizes; (void)n_in; (void)out_size; (void)ws_size;
    const float* features = (const float*)d_in[0];
    const float* gfeat    = (const float*)d_in[1];
    const float* b1       = (const float*)d_in[3];
    const float* b2       = (const float*)d_in[5];
    _Float16* wp = (_Float16*)d_ws;

    struct WSpec { int idx; int K; int N; size_t off; };
    const WSpec specs[16] = {
        { 2, 512, 512, OFF_W1    },  // in_W1
        { 4, 512, 256, OFF_W2    },  // in_W2
        { 6, 128,  64, OFF_FW    },  // fw_W
        { 7,  64,  64, OFF_FWGLU },  // fw_glu_W
        { 8, 128, 192, OFF_WIH1  },  // gru1_Wih
        { 9,  64, 192, OFF_WHH1  },  // gru1_Whh
        {10,  64,  64, OFF_GLU1  },  // glu1_W
        {11, 128, 192, OFF_WIH2  },  // gru2_Wih
        {12,  64, 192, OFF_WHH2  },  // gru2_Whh
        {13,  64,  64, OFF_GLU2  },  // glu2_W
        {14, 128, 192, OFF_WIH3  },  // gru3_Wih
        {15,  64, 192, OFF_WHH3  },  // gru3_Whh
        {16,  64,  64, OFF_GLU3  },  // glu3_W
        {17, 320, 128, OFF_SKIPD },  // skip_dense_W
        {18, 128, 128, OFF_SKIPG },  // skip_glu_W
        {19, 128,  64, OFF_OUTW  },  // out_W
    };
    for (int i = 0; i < 16; ++i) {
        int total  = specs[i].K * specs[i].N;
        int blocks = (total + 255) / 256;
        fargan_pack_w<<<blocks, 256, 0, stream>>>(
            (const float*)d_in[specs[i].idx], wp + specs[i].off,
            specs[i].K, specs[i].N);
    }

    // 4 workgroups x 16 batch rows each; 128 threads = 4 wave32s
    fargan_main<<<4, NTHREADS, 0, stream>>>(features, gfeat, b1, b2, wp, (float*)d_out);
}
